// Glm4Attention_32504312496658
// MI455X (gfx1250) — compile-verified, matrix operand order checked by reading, not run on hardware
//
#include <hip/hip_runtime.h>
#include <hip/hip_bf16.h>

// ---------------------------------------------------------------------------
// GLM4-style attention block for MI455X (gfx1250), wave32 + WMMA bf16.
// - all matmuls via v_wmma_f32_16x16x32_bf16 (f32 accumulate)
// - all LDS staging via global_load_async_to_lds_b128 (ASYNCcnt pipeline)
// - weights pre-swizzled [K/32][N][32], V pre-transposed [kv*D][L] so every
//   staging transfer is a straight 16B copy (no register transposes in loops)
// ---------------------------------------------------------------------------

typedef __attribute__((ext_vector_type(16))) __bf16 v16bf;
typedef __attribute__((ext_vector_type(8)))  float  v8f;

#define L_SEQ   2048
#define HID_DIM 4096
#define NH      32
#define NKV     2
#define DH      128
#define REP_LG2 4              // H/KV = 16 -> shift 4
#define ROT_H   32             // ROT/2
#define SCALE_F 0.08838834764831845f

#define BK       32            // GEMM K-tile
#define BSTRIDE  40            // padded LDS stride (bf16) for A & B tiles
#define VSTRIDE  72            // padded LDS stride (bf16) for transposed V

static __device__ __forceinline__ v8f wmma_bf16(v16bf a, v16bf b, v8f c) {
    return __builtin_amdgcn_wmma_f32_16x16x32_bf16(false, a, false, b,
                                                   (short)0, c, false, false);
}

// async 16B copy: global -> LDS, tracked by ASYNCcnt (CDNA5)
static __device__ __forceinline__ void async_b128(unsigned lds_off,
                                                  const void* gaddr) {
    asm volatile("global_load_async_to_lds_b128 %0, %1, off"
                 :: "v"(lds_off), "v"(gaddr) : "memory");
}
static __device__ __forceinline__ void wait_async() {
    asm volatile("s_wait_asynccnt 0x0" ::: "memory");
}
static __device__ __forceinline__ unsigned lds_addr(const void* p) {
    return (unsigned)(size_t)p;   // LDS aperture: low 32 bits = LDS offset
}

// ------------------------------ fp32 -> bf16 -------------------------------
__global__ void cvt_f32_bf16(const float* __restrict__ in,
                             __bf16* __restrict__ out, int n) {
    int i = blockIdx.x * blockDim.x + threadIdx.x;
    if (i < n) out[i] = (__bf16)in[i];
}

// fp32 [K][N] -> bf16 swizzled [K/32][N][32]  (one-time weight repack)
__global__ void cvt_swizzle_w(const float* __restrict__ in,
                              __bf16* __restrict__ out, int K, int N) {
    int i = blockIdx.x * blockDim.x + threadIdx.x;
    if (i >= K * N) return;
    int n = i % N, k = i / N;
    out[((size_t)(k >> 5) * N + n) * 32 + (k & 31)] = (__bf16)in[i];
}

// ------------------------------- WMMA GEMM ---------------------------------
// C[M,N] = A[M,K] @ W[K,N] (+ bias). BM=128, BN=128, BK=32, double-buffered.
// A row-major; W pre-swizzled [K/32][N][32]. LDS: A [m][k], B [n][k], both
// stride 40 -> every fragment is 2x ds_load_b128; staging is pure async copy.
__global__ __launch_bounds__(256)
void gemm_bf16(const __bf16* __restrict__ A, const __bf16* __restrict__ Bw,
               const float* __restrict__ bias, __bf16* __restrict__ outb,
               float* __restrict__ outf, int M, int N, int K) {
    __shared__ __align__(16) __bf16 As[2][128 * BSTRIDE];
    __shared__ __align__(16) __bf16 Bs[2][128 * BSTRIDE];

    const int tid   = threadIdx.x;
    const int wid   = tid >> 5;
    const int lane  = tid & 31;
    const int wm    = wid >> 2;        // 0..1
    const int wn    = wid & 3;         // 0..3
    const int laneM = lane & 15;
    const int laneH = lane >> 4;
    const int m0 = blockIdx.y * 128;
    const int n0 = blockIdx.x * 128;

    v8f acc[4][2];
    const v8f vzero = {0.f, 0.f, 0.f, 0.f, 0.f, 0.f, 0.f, 0.f};
#pragma unroll
    for (int sm = 0; sm < 4; ++sm)
#pragma unroll
        for (int sn = 0; sn < 2; ++sn) acc[sm][sn] = vzero;

    // each thread stages 2x A-chunks + 2x B-chunks (16B each) per K-step
    auto stage = [&](int k0, int buf) {
#pragma unroll
        for (int cc = 0; cc < 2; ++cc) {
            int c = tid + cc * 256;            // 0..511
            int r = c >> 2, col = (c & 3) * 8; // r: A-row / B-col, 4x8 lanes
            async_b128(lds_addr(&As[buf][r * BSTRIDE + col]),
                       &A[(size_t)(m0 + r) * K + k0 + col]);
            async_b128(lds_addr(&Bs[buf][r * BSTRIDE + col]),
                       &Bw[((size_t)(k0 >> 5) * N + n0 + r) * 32 + col]);
        }
    };

    const int nk = K / BK;
    stage(0, 0);
    wait_async();
    __syncthreads();

    for (int kt = 0; kt < nk; ++kt) {
        if (kt + 1 < nk) stage((kt + 1) * BK, (kt + 1) & 1);
        const int buf = kt & 1;

        v16bf af[4];
#pragma unroll
        for (int sm = 0; sm < 4; ++sm) {
            int row = wm * 64 + sm * 16 + laneM;
            union { v16bf v; unsigned u[8]; } fa;
#pragma unroll
            for (int i = 0; i < 8; ++i) {
                int kk = (i >> 2) * 16 + laneH * 8 + (i & 3) * 2;
                fa.u[i] = *(const unsigned*)&As[buf][row * BSTRIDE + kk];
            }
            af[sm] = fa.v;
        }
        v16bf bfr[2];
#pragma unroll
        for (int sn = 0; sn < 2; ++sn) {
            int col = wn * 32 + sn * 16 + laneM;
            union { v16bf v; unsigned u[8]; } fb;
#pragma unroll
            for (int i = 0; i < 8; ++i) {
                int kk = (i >> 2) * 16 + laneH * 8 + (i & 3) * 2;
                fb.u[i] = *(const unsigned*)&Bs[buf][col * BSTRIDE + kk];
            }
            bfr[sn] = fb.v;
        }
#pragma unroll
        for (int sm = 0; sm < 4; ++sm)
#pragma unroll
            for (int sn = 0; sn < 2; ++sn)
                acc[sm][sn] = wmma_bf16(af[sm], bfr[sn], acc[sm][sn]);

        wait_async();          // prefetch for (kt+1) landed
        __syncthreads();
    }

    // epilogue: C layout VGPR j -> row j (lanes 0-15) / j+8 (lanes 16-31)
#pragma unroll
    for (int sm = 0; sm < 4; ++sm)
#pragma unroll
        for (int sn = 0; sn < 2; ++sn)
#pragma unroll
            for (int j = 0; j < 8; ++j) {
                int r = m0 + wm * 64 + sm * 16 + j + 8 * laneH;
                int c = n0 + wn * 32 + sn * 16 + laneM;
                float v = acc[sm][sn][j];
                if (bias) v += bias[c];
                if (outb) outb[(size_t)r * N + c] = (__bf16)v;
                else      outf[(size_t)r * N + c] = v;
            }
}

// --------------------------- RoPE + head repack ----------------------------
// in: [L][nh*D] bf16 -> out: [nh][L][D] bf16, half-rotation on first 64 dims.
__global__ void rope_pack(const __bf16* __restrict__ in,
                          __bf16* __restrict__ out,
                          const float* __restrict__ cs,
                          const float* __restrict__ sn,
                          int nh, int do_rope) {
    int i = blockIdx.x * blockDim.x + threadIdx.x;
    int total = nh * L_SEQ * DH;
    if (i >= total) return;
    int d = i % DH;
    int l = (i / DH) % L_SEQ;
    int h = i / (DH * L_SEQ);
    const __bf16* row = in + (size_t)l * (nh * DH) + (size_t)h * DH;
    float v;
    if (do_rope && d < 64) {
        if (d < 32) {
            float x1 = (float)row[d], x2 = (float)row[d + 32];
            float c = cs[l * ROT_H + d], s = sn[l * ROT_H + d];
            v = x1 * c - x2 * s;
        } else {
            int dd = d - 32;
            float x1 = (float)row[dd], x2 = (float)row[d];
            float c = cs[l * ROT_H + dd], s = sn[l * ROT_H + dd];
            v = x2 * c + x1 * s;
        }
    } else {
        v = (float)row[d];
    }
    out[(size_t)h * L_SEQ * DH + (size_t)l * DH + d] = (__bf16)v;
}

// V: [L][kv*D] -> transposed [kv*D][L]  (makes attention V tiles contiguous)
__global__ void pack_v_t(const __bf16* __restrict__ in,
                         __bf16* __restrict__ out) {
    int i = blockIdx.x * blockDim.x + threadIdx.x;  // over KV*D*L, l fastest
    if (i >= NKV * DH * L_SEQ) return;
    int l  = i % L_SEQ;
    int hd = i / L_SEQ;                              // h*DH + d
    out[i] = in[(size_t)l * (NKV * DH) + hd];
}

// ---------------------- flash attention (causal, GQA) ----------------------
// grid = (L/64, H); block = 256 (8 waves, 4x2). 64 query rows per block.
// K tile row-major [key][d]; V tile [d][key] copied from global-transposed V.
// All staging via async copies.
__global__ __launch_bounds__(256)
void attn_kernel(const __bf16* __restrict__ Q,   // [H][L][D]
                 const __bf16* __restrict__ Kc,  // [KV][L][D]
                 const __bf16* __restrict__ Vt_g,// [KV*D][L]
                 __bf16* __restrict__ ctx) {     // [L][H*D]
    __shared__ __align__(16) __bf16 Ks[64 * 128];          // 16 KB
    __shared__ __align__(16) __bf16 Vt[128 * VSTRIDE];     // 18 KB
    __shared__ __align__(16) float  Sbuf[64 * 64];         // 16 KB
    __shared__ __align__(16) __bf16 Pbuf[64 * 64];         //  8 KB
    __shared__ float rowfac[64];

    const int tid   = threadIdx.x;
    const int wid   = tid >> 5;
    const int lane  = tid & 31;
    const int wm    = wid >> 1;        // 0..3 : 16-row group
    const int wn    = wid & 1;         // 0..1
    const int laneM = lane & 15;
    const int laneH = lane >> 4;
    const int head  = blockIdx.y;
    const int kvh   = head >> REP_LG2;
    const int qt    = blockIdx.x;
    const int qbase = qt * 64;

    // Q fragments loaded once (rows wm*16.., full D=128 => 4 fragments)
    v16bf qf[4];
    {
        const __bf16* qrow =
            Q + ((size_t)head * L_SEQ + qbase + wm * 16 + laneM) * DH;
#pragma unroll
        for (int f = 0; f < 4; ++f) {
            union { v16bf v; unsigned u[8]; } fa;
#pragma unroll
            for (int i = 0; i < 8; ++i) {
                int kk = f * 32 + (i >> 2) * 16 + laneH * 8 + (i & 3) * 2;
                fa.u[i] = *(const unsigned*)&qrow[kk];
            }
            qf[f] = fa.v;
        }
    }

    const v8f vzero = {0.f, 0.f, 0.f, 0.f, 0.f, 0.f, 0.f, 0.f};
    v8f o[4];
#pragma unroll
    for (int t = 0; t < 4; ++t) o[t] = vzero;
    float m_prev = -3.0e38f, lsum = 0.f;   // live only in threads < 64

    for (int kt = 0; kt <= qt; ++kt) {
        const int kbase = kt * 64;
        // K tile: 64x128, 4 chunks/thread; V tile: 128x64, 4 chunks/thread
#pragma unroll
        for (int cc = 0; cc < 4; ++cc) {
            int c  = tid + cc * 256;               // 0..1023
            int kr = c >> 4, kcol = (c & 15) * 8;  // K: row, 16x8 cols
            async_b128(lds_addr(&Ks[kr * 128 + kcol]),
                       &Kc[((size_t)kvh * L_SEQ + kbase + kr) * DH + kcol]);
            int vd = c >> 3, vkey = (c & 7) * 8;   // V: d-row, 8x8 keys
            async_b128(lds_addr(&Vt[vd * VSTRIDE + vkey]),
                       &Vt_g[((size_t)(kvh * DH + vd)) * L_SEQ + kbase + vkey]);
        }
        wait_async();
        __syncthreads();

        // S = Q @ K^T  (wave tile: rows wm*16.., cols wn*32..)
        v8f s0 = vzero, s1 = vzero;
#pragma unroll
        for (int f = 0; f < 4; ++f) {
            union { v16bf v; unsigned u[8]; } fb0, fb1;
            int ncol0 = wn * 32 + laneM;
#pragma unroll
            for (int i = 0; i < 8; ++i) {
                int kk = f * 32 + (i >> 2) * 16 + laneH * 8 + (i & 3) * 2;
                fb0.u[i] = *(const unsigned*)&Ks[ncol0 * 128 + kk];
                fb1.u[i] = *(const unsigned*)&Ks[(ncol0 + 16) * 128 + kk];
            }
            s0 = wmma_bf16(qf[f], fb0.v, s0);
            s1 = wmma_bf16(qf[f], fb1.v, s1);
        }
        // scale + causal mask -> Sbuf
#pragma unroll
        for (int j = 0; j < 8; ++j) {
            int r  = wm * 16 + j + 8 * laneH;
            int gq = qbase + r;
            int c0 = wn * 32 + laneM;
            float v0 = s0[j] * SCALE_F, v1 = s1[j] * SCALE_F;
            if (kbase + c0 > gq)      v0 = -1.0e30f;
            if (kbase + c0 + 16 > gq) v1 = -1.0e30f;
            Sbuf[r * 64 + c0]      = v0;
            Sbuf[r * 64 + c0 + 16] = v1;
        }
        __syncthreads();

        // online softmax: one thread per query row
        if (tid < 64) {
            float mx = -3.0e38f;
#pragma unroll 8
            for (int j = 0; j < 64; ++j) mx = fmaxf(mx, Sbuf[tid * 64 + j]);
            float mnew = fmaxf(m_prev, mx);
            float fac  = __expf(m_prev - mnew);
            float s = 0.f;
#pragma unroll 8
            for (int j = 0; j < 64; ++j) {
                float e = __expf(Sbuf[tid * 64 + j] - mnew);
                s += e;
                Pbuf[tid * 64 + j] = (__bf16)e;
            }
            lsum = lsum * fac + s;
            m_prev = mnew;
            rowfac[tid] = fac;
        }
        __syncthreads();

        // rescale running O by row factors
        float fj[8];
#pragma unroll
        for (int j = 0; j < 8; ++j)
            fj[j] = rowfac[wm * 16 + j + 8 * laneH];
#pragma unroll
        for (int t = 0; t < 4; ++t)
#pragma unroll
            for (int j = 0; j < 8; ++j) o[t][j] *= fj[j];

        // O += P @ V  (wave cols: wn*64 .. +64 -> 4 subtiles)
#pragma unroll
        for (int half = 0; half < 2; ++half) {
            union { v16bf v; unsigned u[8]; } fp;
            int prow = wm * 16 + laneM;
#pragma unroll
            for (int i = 0; i < 8; ++i) {
                int kk = half * 32 + (i >> 2) * 16 + laneH * 8 + (i & 3) * 2;
                fp.u[i] = *(const unsigned*)&Pbuf[prow * 64 + kk];
            }
#pragma unroll
            for (int t = 0; t < 4; ++t) {
                int col = wn * 64 + t * 16 + laneM;    // d-index
                union { v16bf v; unsigned u[8]; } fv;
#pragma unroll
                for (int i = 0; i < 8; ++i) {
                    int kk = half * 32 + (i >> 2) * 16 + laneH * 8 + (i & 3) * 2;
                    fv.u[i] = *(const unsigned*)&Vt[col * VSTRIDE + kk];
                }
                o[t] = wmma_bf16(fp.v, fv.v, o[t]);
            }
        }
        __syncthreads();
    }

    // final normalization + store context (bf16, [L][H*D])
    if (tid < 64) rowfac[tid] = 1.0f / lsum;
    __syncthreads();
    float fj[8];
#pragma unroll
    for (int j = 0; j < 8; ++j) fj[j] = rowfac[wm * 16 + j + 8 * laneH];
#pragma unroll
    for (int t = 0; t < 4; ++t)
#pragma unroll
        for (int j = 0; j < 8; ++j) {
            int r = qbase + wm * 16 + j + 8 * laneH;
            int c = head * DH + wn * 64 + t * 16 + laneM;
            ctx[(size_t)r * (NH * DH) + c] = (__bf16)(o[t][j] * fj[j]);
        }
}

// ------------------------------- launcher ----------------------------------
extern "C" void kernel_launch(void* const* d_in, const int* in_sizes, int n_in,
                              void* d_out, int out_size, void* d_ws,
                              size_t ws_size, hipStream_t stream) {
    (void)in_sizes; (void)n_in; (void)out_size; (void)ws_size;

    const float* x   = (const float*)d_in[0];
    const float* Wq  = (const float*)d_in[1];
    const float* bq  = (const float*)d_in[2];
    const float* Wk  = (const float*)d_in[3];
    const float* bk  = (const float*)d_in[4];
    const float* Wv  = (const float*)d_in[5];
    const float* bv  = (const float*)d_in[6];
    const float* Wo  = (const float*)d_in[7];
    const float* cs  = (const float*)d_in[8];
    const float* sn  = (const float*)d_in[9];
    // d_in[10] = attention_mask (pure causal; applied analytically)

    char* ws = (char*)d_ws;
    size_t off = 0;
    auto carve = [&](size_t bytes) {
        void* p = ws + off;
        off = (off + bytes + 255) & ~(size_t)255;
        return p;
    };

    const size_t nX  = (size_t)L_SEQ * HID_DIM;         // 8.39M
    const size_t nWq = (size_t)HID_DIM * HID_DIM;       // 16.8M
    const size_t nWk = (size_t)HID_DIM * NKV * DH;      // 1.05M
    const size_t nKV = (size_t)L_SEQ * NKV * DH;

    __bf16* xb    = (__bf16*)carve(nX  * 2);
    __bf16* Wqb   = (__bf16*)carve(nWq * 2);            // swizzled
    __bf16* Wkb   = (__bf16*)carve(nWk * 2);            // swizzled
    __bf16* Wvb   = (__bf16*)carve(nWk * 2);            // swizzled
    __bf16* Wob   = (__bf16*)carve(nWq * 2);            // swizzled
    __bf16* q_tmp = (__bf16*)carve(nX  * 2);            // [L][H*D]
    __bf16* k_tmp = (__bf16*)carve(nKV * 2);
    __bf16* v_tmp = (__bf16*)carve(nKV * 2);
    __bf16* qh    = (__bf16*)carve(nX  * 2);            // [H][L][D]
    __bf16* kh    = (__bf16*)carve(nKV * 2);            // [KV][L][D]
    __bf16* vt    = (__bf16*)carve(nKV * 2);            // [KV*D][L]
    __bf16* ctx   = q_tmp;  // q_tmp is dead after rope_pack -> reuse for ctx

    // 1) convert / repack inputs to bf16
    cvt_f32_bf16<<<(int)((nX + 255) / 256), 256, 0, stream>>>(x, xb, (int)nX);
    cvt_swizzle_w<<<(int)((nWq + 255) / 256), 256, 0, stream>>>(
        Wq, Wqb, HID_DIM, HID_DIM);
    cvt_swizzle_w<<<(int)((nWk + 255) / 256), 256, 0, stream>>>(
        Wk, Wkb, HID_DIM, NKV * DH);
    cvt_swizzle_w<<<(int)((nWk + 255) / 256), 256, 0, stream>>>(
        Wv, Wvb, HID_DIM, NKV * DH);
    cvt_swizzle_w<<<(int)((nWq + 255) / 256), 256, 0, stream>>>(
        Wo, Wob, HID_DIM, HID_DIM);

    // 2) QKV projections (WMMA GEMM + bias)
    gemm_bf16<<<dim3(HID_DIM / 128, L_SEQ / 128), 256, 0, stream>>>(
        xb, Wqb, bq, q_tmp, nullptr, L_SEQ, HID_DIM, HID_DIM);
    gemm_bf16<<<dim3((NKV * DH) / 128, L_SEQ / 128), 256, 0, stream>>>(
        xb, Wkb, bk, k_tmp, nullptr, L_SEQ, NKV * DH, HID_DIM);
    gemm_bf16<<<dim3((NKV * DH) / 128, L_SEQ / 128), 256, 0, stream>>>(
        xb, Wvb, bv, v_tmp, nullptr, L_SEQ, NKV * DH, HID_DIM);

    // 3) RoPE + repack (Q,K head-major; V globally transposed)
    {
        int tq = NH * L_SEQ * DH, tk = NKV * L_SEQ * DH;
        rope_pack<<<(tq + 255) / 256, 256, 0, stream>>>(q_tmp, qh, cs, sn, NH, 1);
        rope_pack<<<(tk + 255) / 256, 256, 0, stream>>>(k_tmp, kh, cs, sn, NKV, 1);
        pack_v_t <<<(tk + 255) / 256, 256, 0, stream>>>(v_tmp, vt);
    }

    // 4) causal flash attention (WMMA for QK^T and PV)
    attn_kernel<<<dim3(L_SEQ / 64, NH), 256, 0, stream>>>(qh, kh, vt, ctx);

    // 5) output projection -> f32 d_out
    gemm_bf16<<<dim3(HID_DIM / 128, L_SEQ / 128), 256, 0, stream>>>(
        ctx, Wob, nullptr, nullptr, (float*)d_out, L_SEQ, HID_DIM, HID_DIM);
}
